// TransformerBlock_69166153334830
// MI455X (gfx1250) — matrix operand order repaired and checked by reading, not compiled, not run on hardware
//
#include <hip/hip_runtime.h>
#include <math.h>

#define D_MODELC 1024
#define S_LENC   2048
#define NTOKC    4096
#define N_HEADSC 16
#define D_KC     64
#define D_FFC    4096
#define KSTEP    64

typedef __attribute__((ext_vector_type(16))) __bf16 v16bf;
typedef __attribute__((ext_vector_type(8)))  float  v8f;
typedef __attribute__((ext_vector_type(4)))  unsigned int u32x4;
typedef __attribute__((ext_vector_type(8)))  int i32x8;
typedef __attribute__((ext_vector_type(4)))  int i32x4;

#if __has_builtin(__builtin_amdgcn_tensor_load_to_lds)
#define HAVE_TDM 1
#else
#define HAVE_TDM 0
#endif

union Frag16 {
  v16bf v;
  unsigned short u[16];
  uint4 q4[2];
};

__device__ __forceinline__ unsigned short f2bf(float f) {
  union { float f; unsigned u; } x; x.f = f;
  unsigned r = x.u + 0x7fffu + ((x.u >> 16) & 1u);
  return (unsigned short)(r >> 16);
}

__device__ __forceinline__ v8f wmma_bf16(const Frag16& a, const Frag16& b, v8f c) {
  return __builtin_amdgcn_wmma_f32_16x16x32_bf16(false, a.v, false, b.v, (short)0, c,
                                                 false, false);
}

#if HAVE_TDM
// TDM: DMA a [rows=128][cols=KSTEP] bf16 tile (row stride K elements) from
// global into LDS at lds_off, packed row-major. Descriptor per CDNA5 ISA §8.
__device__ __forceinline__ void tdm_load_b_tile(const unsigned short* gsrc,
                                                unsigned lds_off, int K, int N) {
  unsigned long long ga = (unsigned long long)(uintptr_t)gsrc;
  u32x4 g0;
  g0[0] = 1u;                                     // count=1, user mode
  g0[1] = lds_off;                                // lds_addr (bytes)
  g0[2] = (unsigned)(ga & 0xffffffffu);           // global_addr[31:0]
  g0[3] = (unsigned)((ga >> 32) & 0x1ffffffu) | (2u << 30);  // addr[56:32], type=2
  i32x8 g1;
  g1[0] = 0x00010000;                             // data_size=1 (2 bytes), mask=0
  g1[1] = (int)(((unsigned)K & 0xffffu) << 16);   // tensor_dim0 lo16 in [63:48]
  g1[2] = (int)((((unsigned)K >> 16) & 0xffffu) | (((unsigned)N & 0xffffu) << 16));
  g1[3] = (int)((((unsigned)N >> 16) & 0xffffu) | ((unsigned)KSTEP << 16));  // tile_dim0
  g1[4] = 128;                                    // tile_dim1=128, tile_dim2=0
  g1[5] = (int)(unsigned)K;                       // tensor_dim0_stride lo32
  g1[6] = 0;                                      // stride hi16, dim1_stride lo16
  g1[7] = 0;
  i32x4 z4 = {0, 0, 0, 0};
#if __has_include(<hip/amd_detail/amd_gfx1250_TDM.h>)
  i32x8 z8 = {0, 0, 0, 0, 0, 0, 0, 0};
  __builtin_amdgcn_tensor_load_to_lds(g0, g1, z4, z4, z8, 0);
#else
  __builtin_amdgcn_tensor_load_to_lds(g0, g1, z4, z4, 0);
#endif
}
#endif

// ---------------------------------------------------------------------------
// fp32 [K][N] -> bf16 transposed [N][K]
// ---------------------------------------------------------------------------
__global__ __launch_bounds__(256) void cvt_transpose(const float* __restrict__ w,
                                                     unsigned short* __restrict__ wt,
                                                     int K, int N) {
  int idx = blockIdx.x * 256 + threadIdx.x;
  int k = idx / N, n = idx - k * N;
  wt[(size_t)n * K + k] = f2bf(w[idx]);
}

// ---------------------------------------------------------------------------
// LayerNorm over D=1024, fp32 in -> bf16 out. One block (256 thr) per token.
// ---------------------------------------------------------------------------
__global__ __launch_bounds__(256) void layernorm_bf16(const float* __restrict__ x,
                                                      const float* __restrict__ gamma,
                                                      const float* __restrict__ beta,
                                                      unsigned short* __restrict__ out) {
  __shared__ float sh_s[8], sh_s2[8];
  const int t = blockIdx.x, tid = threadIdx.x;
  const float* row = x + (size_t)t * D_MODELC;
  float v[4], s = 0.f, s2 = 0.f;
#pragma unroll
  for (int j = 0; j < 4; ++j) {
    v[j] = row[tid + j * 256];
    s += v[j];
    s2 += v[j] * v[j];
  }
#pragma unroll
  for (int off = 16; off; off >>= 1) {
    s += __shfl_xor(s, off, 32);
    s2 += __shfl_xor(s2, off, 32);
  }
  if ((tid & 31) == 0) { sh_s[tid >> 5] = s; sh_s2[tid >> 5] = s2; }
  __syncthreads();
  s = 0.f; s2 = 0.f;
#pragma unroll
  for (int i = 0; i < 8; ++i) { s += sh_s[i]; s2 += sh_s2[i]; }
  const float mu = s * (1.f / D_MODELC);
  const float var = s2 * (1.f / D_MODELC) - mu * mu;
  const float rs = rsqrtf(var + 1e-5f);
#pragma unroll
  for (int j = 0; j < 4; ++j) {
    int i = tid + j * 256;
    out[(size_t)t * D_MODELC + i] = f2bf((v[j] - mu) * rs * gamma[i] + beta[i]);
  }
}

// ---------------------------------------------------------------------------
// WMMA GEMM: C[M=grid.y*128][N] = A[M][K](bf16) x Bt[N][K](bf16, pre-transposed)
// 256 threads = 8 waves; block tile 128x128; wave tile 32x64; K-step 64.
// Double-buffered LDS; B tile staged by the Tensor Data Mover when available
// (issued by wave 0, overlapped with WMMA, fenced by s_wait_tensorcnt).
// A tile staged pre-swizzled into the CDNA5 16x32 A-fragment layout.
// ---------------------------------------------------------------------------
enum { EPI_HEADQK = 0, EPI_HEADV = 1, EPI_GELU = 2, EPI_RES = 3 };

template <int EPI>
__global__ __launch_bounds__(256) void gemm_bf16_128x128(
    const unsigned short* __restrict__ A,
    const unsigned short* __restrict__ Bt,
    const float* __restrict__ bias,
    const float* __restrict__ res,
    unsigned short* __restrict__ out_bf,
    float* __restrict__ out_f,
    int N, int K) {
  __shared__ __align__(32) unsigned short lA[2][2][8][32][16];  // 32 KB frag-layout
  __shared__ __align__(32) unsigned short lB[2][128][KSTEP];    // 32 KB [n][k]
  const int tid = threadIdx.x;
  const int lane = tid & 31, wid = tid >> 5;
  const int wm = wid & 3, wn = wid >> 2;
  const int mBase = blockIdx.y * 128, nBase = blockIdx.x * 128;
  const int row2 = tid >> 1, half = tid & 1;

  const v8f zero = {0.f, 0.f, 0.f, 0.f, 0.f, 0.f, 0.f, 0.f};
  v8f acc[2][4];
#pragma unroll
  for (int mi = 0; mi < 2; ++mi)
#pragma unroll
    for (int nt = 0; nt < 4; ++nt) acc[mi][nt] = zero;

  const unsigned short* aSrc = A + (size_t)(mBase + row2) * K + half * 16;
#if HAVE_TDM
  const unsigned short* bTile = Bt + (size_t)nBase * K;
#else
  const unsigned short* bSrc = Bt + (size_t)(nBase + row2) * K + half * 32;
#endif
  const int nsteps = K / KSTEP;

  auto stageA = [&](int s, int buf) {
    const int k0 = s * KSTEP;
#pragma unroll
    for (int kc = 0; kc < 2; ++kc) {
      uint4 a0 = *(const uint4*)(aSrc + k0 + kc * 32);      // k' = half*16 + 0..7
      uint4 a1 = *(const uint4*)(aSrc + k0 + kc * 32 + 8);  // k' = half*16 + 8..15
      *(uint4*)&lA[buf][kc][row2 >> 4][(row2 & 15) + 16 * half][0] = a0;
      *(uint4*)&lA[buf][kc][row2 >> 4][(row2 & 15) + 16 * half][8] = a1;
    }
  };
  auto stageB = [&](int s, int buf) {
    const int k0 = s * KSTEP;
#if HAVE_TDM
    if (wid == 0)
      tdm_load_b_tile(bTile + k0, (unsigned)(uintptr_t)&lB[buf][0][0], K, N);
#else
#pragma unroll
    for (int q = 0; q < 2; ++q) {
      uint4 b0 = *(const uint4*)(bSrc + k0 + q * 16);
      uint4 b1 = *(const uint4*)(bSrc + k0 + q * 16 + 8);
      *(uint4*)&lB[buf][row2][half * 32 + q * 16] = b0;
      *(uint4*)&lB[buf][row2][half * 32 + q * 16 + 8] = b1;
    }
#endif
  };

  stageA(0, 0);
  stageB(0, 0);
#if HAVE_TDM
  if (wid == 0) __builtin_amdgcn_s_wait_tensorcnt((short)0);
#endif
  __syncthreads();

  for (int s = 0; s < nsteps; ++s) {
    const int buf = s & 1;
    if (s + 1 < nsteps) {  // prefetch next tile while computing this one
      stageA(s + 1, buf ^ 1);
      stageB(s + 1, buf ^ 1);
    }
#pragma unroll
    for (int kc = 0; kc < 2; ++kc) {
      Frag16 a[2], b[4];
      a[0].v = *(const v16bf*)&lA[buf][kc][2 * wm][lane][0];
      a[1].v = *(const v16bf*)&lA[buf][kc][2 * wm + 1][lane][0];
#pragma unroll
      for (int nt = 0; nt < 4; ++nt)
        b[nt].v = *(const v16bf*)
            &lB[buf][wn * 64 + nt * 16 + (lane & 15)][kc * 32 + (lane >> 4) * 16];
#pragma unroll
      for (int mi = 0; mi < 2; ++mi)
#pragma unroll
        for (int nt = 0; nt < 4; ++nt) acc[mi][nt] = wmma_bf16(a[mi], b[nt], acc[mi][nt]);
    }
#if HAVE_TDM
    if (wid == 0 && s + 1 < nsteps) __builtin_amdgcn_s_wait_tensorcnt((short)0);
#endif
    __syncthreads();
  }

  // --- epilogue
  const int g = lane >> 4, c = lane & 15;
#pragma unroll
  for (int mi = 0; mi < 2; ++mi) {
#pragma unroll
    for (int nt = 0; nt < 4; ++nt) {
      const int col = nBase + wn * 64 + nt * 16 + c;
      const float bs = bias[col];
#pragma unroll
      for (int r = 0; r < 8; ++r) {
        const int row = mBase + wm * 32 + mi * 16 + r + 8 * g;
        float val = acc[mi][nt][r] + bs;
        if (EPI == EPI_HEADQK) {
          const int b_ = row >> 11, s_ = row & 2047, h_ = col >> 6, d_ = col & 63;
          out_bf[((size_t)(b_ * N_HEADSC + h_) * S_LENC + s_) * D_KC + d_] = f2bf(val);
        } else if (EPI == EPI_HEADV) {
          const int b_ = row >> 11, s_ = row & 2047, h_ = col >> 6, d_ = col & 63;
          out_bf[((size_t)(b_ * N_HEADSC + h_) * D_KC + d_) * S_LENC + s_] = f2bf(val);
        } else if (EPI == EPI_GELU) {
          val = 0.5f * val * (1.f + erff(val * 0.70710678118654752f));
          out_bf[(size_t)row * N + col] = f2bf(val);
        } else {  // EPI_RES
          out_f[(size_t)row * N + col] = val + res[(size_t)row * N + col];
        }
      }
    }
  }
}

// ---------------------------------------------------------------------------
// Flash attention: one wave per 16-row q tile; online softmax; causal.
// Q,K: [BH][S][64] bf16   V: [BH][64][S] bf16 (transposed)
// out: [B*S][1024] bf16
// ---------------------------------------------------------------------------
__global__ __launch_bounds__(256) void attention_fa(
    const unsigned short* __restrict__ Q,
    const unsigned short* __restrict__ Kh_,
    const unsigned short* __restrict__ Vt,
    unsigned short* __restrict__ attn,
    const int* __restrict__ maskp) {
  __shared__ __align__(32) unsigned short lp[8][32][16];  // per-wave P bounce
  const int lane = threadIdx.x & 31;
  const int wid = threadIdx.x >> 5;
  const int g = lane >> 4;
  const int c = lane & 15;
  const int qt = blockIdx.x * 8 + wid;
  const int bh = blockIdx.y;
  const int causal = maskp[0];

  const unsigned short* Qh = Q + (size_t)bh * S_LENC * D_KC;
  const unsigned short* Kh = Kh_ + (size_t)bh * S_LENC * D_KC;
  const unsigned short* Vh = Vt + (size_t)bh * D_KC * S_LENC;

  // Q fragments (A layout): d = dc*32 + 16*g + {0..15}
  Frag16 aq0, aq1;
  {
    const unsigned short* qr = Qh + (size_t)(qt * 16 + c) * D_KC;
    const int d0 = 16 * g;
    aq0.q4[0] = *(const uint4*)(qr + d0);
    aq0.q4[1] = *(const uint4*)(qr + d0 + 8);
    aq1.q4[0] = *(const uint4*)(qr + 32 + d0);
    aq1.q4[1] = *(const uint4*)(qr + 32 + d0 + 8);
  }

  const v8f zero = {0.f, 0.f, 0.f, 0.f, 0.f, 0.f, 0.f, 0.f};
  v8f o[4];
#pragma unroll
  for (int j = 0; j < 4; ++j) o[j] = zero;
  float mrow[8], lrow[8];
#pragma unroll
  for (int r = 0; r < 8; ++r) { mrow[r] = -1e30f; lrow[r] = 0.f; }

  const int kend = causal ? (qt * 16 + 16) : S_LENC;
  for (int kc = 0; kc < kend; kc += 32) {
    // K fragments (B layout): lane holds key = base + c, contiguous d
    Frag16 b00, b01, b10, b11;
    const unsigned short* kr0 = Kh + (size_t)(kc + c) * D_KC + g * 16;
    const unsigned short* kr1 = Kh + (size_t)(kc + 16 + c) * D_KC + g * 16;
    b00.v = *(const v16bf*)kr0;
    b01.v = *(const v16bf*)(kr0 + 32);
    b10.v = *(const v16bf*)kr1;
    b11.v = *(const v16bf*)(kr1 + 32);
    v8f s0 = wmma_bf16(aq0, b00, wmma_bf16(aq1, b01, zero));
    v8f s1 = wmma_bf16(aq0, b10, wmma_bf16(aq1, b11, zero));

#pragma unroll
    for (int r = 0; r < 8; ++r) {
      float v0 = s0[r] * 0.125f, v1 = s1[r] * 0.125f;
      const int rowq = qt * 16 + r + 8 * g;
      if (causal) {
        if (kc + c > rowq) v0 = -1e30f;
        if (kc + 16 + c > rowq) v1 = -1e30f;
      }
      float mx = fmaxf(v0, v1);
#pragma unroll
      for (int off = 1; off < 16; off <<= 1) mx = fmaxf(mx, __shfl_xor(mx, off, 16));
      const float mn = fmaxf(mrow[r], mx);
      const float al = __expf(mrow[r] - mn);
      const float p0 = __expf(v0 - mn);
      const float p1 = __expf(v1 - mn);
      float sm = p0 + p1;
#pragma unroll
      for (int off = 1; off < 16; off <<= 1) sm += __shfl_xor(sm, off, 16);
      lrow[r] = lrow[r] * al + sm;
      mrow[r] = mn;
      o[0][r] *= al; o[1][r] *= al; o[2][r] *= al; o[3][r] *= al;
      // scatter P into A-fragment layout in LDS
      const int rowl = r + 8 * g;
      const int lane_a = rowl;
      lp[wid][lane_a][c] = f2bf(p0);
      lp[wid][lane_a + 16][c] = f2bf(p1);
    }
    asm volatile("s_wait_dscnt 0" ::: "memory");  // cross-lane LDS RAW within wave
    Frag16 ap;
    ap.v = *(const v16bf*)&lp[wid][lane][0];
#pragma unroll
    for (int j = 0; j < 4; ++j) {
      Frag16 bv;
      bv.v = *(const v16bf*)(Vh + (size_t)(j * 16 + c) * S_LENC + kc + g * 16);
      o[j] = wmma_bf16(ap, bv, o[j]);
    }
  }

  const int b_ = bh >> 4, h_ = bh & 15;
#pragma unroll
  for (int j = 0; j < 4; ++j) {
#pragma unroll
    for (int r = 0; r < 8; ++r) {
      const int srow = qt * 16 + r + 8 * g;
      const float val = o[j][r] / lrow[r];
      attn[(size_t)(b_ * S_LENC + srow) * D_MODELC + h_ * D_KC + j * 16 + c] = f2bf(val);
    }
  }
}

// ---------------------------------------------------------------------------
extern "C" void kernel_launch(void* const* d_in, const int* in_sizes, int n_in,
                              void* d_out, int out_size, void* d_ws, size_t ws_size,
                              hipStream_t stream) {
  const float* x  = (const float*)d_in[0];
  const float* wq = (const float*)d_in[1];
  const float* bq = (const float*)d_in[2];
  const float* wk = (const float*)d_in[3];
  const float* bk = (const float*)d_in[4];
  const float* wv = (const float*)d_in[5];
  const float* bv = (const float*)d_in[6];
  const float* wo = (const float*)d_in[7];
  const float* bo = (const float*)d_in[8];
  const float* w1 = (const float*)d_in[9];
  const float* b1 = (const float*)d_in[10];
  const float* w2 = (const float*)d_in[11];
  const float* b2 = (const float*)d_in[12];
  const float* g1 = (const float*)d_in[13];
  const float* be1 = (const float*)d_in[14];
  const float* g2 = (const float*)d_in[15];
  const float* be2 = (const float*)d_in[16];
  const int* mask = (const int*)d_in[17];

  char* ws = (char*)d_ws;
  const size_t MB = 1024 * 1024;
  unsigned short* wqT = (unsigned short*)(ws + 0 * MB);    // 2 MB
  unsigned short* wkT = (unsigned short*)(ws + 2 * MB);    // 2 MB
  unsigned short* wvT = (unsigned short*)(ws + 4 * MB);    // 2 MB
  unsigned short* woT = (unsigned short*)(ws + 6 * MB);    // 2 MB
  unsigned short* w1T = (unsigned short*)(ws + 8 * MB);    // 8 MB
  unsigned short* w2T = (unsigned short*)(ws + 16 * MB);   // 8 MB
  unsigned short* nx  = (unsigned short*)(ws + 24 * MB);   // 8 MB (reused for ln2)
  unsigned short* qb  = (unsigned short*)(ws + 32 * MB);   // 8 MB
  unsigned short* kb  = (unsigned short*)(ws + 40 * MB);   // 8 MB
  unsigned short* vb  = (unsigned short*)(ws + 48 * MB);   // 8 MB
  unsigned short* ab  = (unsigned short*)(ws + 56 * MB);   // 8 MB
  float*          x1  = (float*)(ws + 64 * MB);            // 16 MB
  unsigned short* hb  = (unsigned short*)(ws + 80 * MB);   // 32 MB

  // weights -> bf16 transposed
  cvt_transpose<<<(1024 * 1024) / 256, 256, 0, stream>>>(wq, wqT, 1024, 1024);
  cvt_transpose<<<(1024 * 1024) / 256, 256, 0, stream>>>(wk, wkT, 1024, 1024);
  cvt_transpose<<<(1024 * 1024) / 256, 256, 0, stream>>>(wv, wvT, 1024, 1024);
  cvt_transpose<<<(1024 * 1024) / 256, 256, 0, stream>>>(wo, woT, 1024, 1024);
  cvt_transpose<<<(1024 * 4096) / 256, 256, 0, stream>>>(w1, w1T, 1024, 4096);
  cvt_transpose<<<(4096 * 1024) / 256, 256, 0, stream>>>(w2, w2T, 4096, 1024);

  // LN1
  layernorm_bf16<<<NTOKC, 256, 0, stream>>>(x, g1, be1, nx);

  // Q, K, V projections
  gemm_bf16_128x128<EPI_HEADQK><<<dim3(8, 32), 256, 0, stream>>>(
      nx, wqT, bq, nullptr, qb, nullptr, 1024, 1024);
  gemm_bf16_128x128<EPI_HEADQK><<<dim3(8, 32), 256, 0, stream>>>(
      nx, wkT, bk, nullptr, kb, nullptr, 1024, 1024);
  gemm_bf16_128x128<EPI_HEADV><<<dim3(8, 32), 256, 0, stream>>>(
      nx, wvT, bv, nullptr, vb, nullptr, 1024, 1024);

  // attention (flash, causal per mask)
  attention_fa<<<dim3(16, 32), 256, 0, stream>>>(qb, kb, vb, ab, mask);

  // O projection + residual -> x1 (fp32)
  gemm_bf16_128x128<EPI_RES><<<dim3(8, 32), 256, 0, stream>>>(
      ab, woT, bo, x, nullptr, x1, 1024, 1024);

  // LN2
  layernorm_bf16<<<NTOKC, 256, 0, stream>>>(x1, g2, be2, nx);

  // FFN1 + exact GELU
  gemm_bf16_128x128<EPI_GELU><<<dim3(32, 32), 256, 0, stream>>>(
      nx, w1T, b1, nullptr, hb, nullptr, 4096, 1024);

  // FFN2 + residual -> d_out (fp32)
  gemm_bf16_128x128<EPI_RES><<<dim3(8, 32), 256, 0, stream>>>(
      hb, w2T, b2, x1, nullptr, (float*)d_out, 1024, 4096);
}
